// GaussianDecoder_1228360647258
// MI455X (gfx1250) — matrix operand order repaired and checked by reading, not compiled, not run on hardware
//
#include <hip/hip_runtime.h>
#include <math.h>

#define IN_DIM 515
#define KP1    544            // fc1 K padded to 17*32
#define HID    256
#define OUTD   11
#define ZD     64
#define BB     2
#define NN     131072
#define WAVES  4

typedef __attribute__((ext_vector_type(16))) __bf16        v16bf;
typedef __attribute__((ext_vector_type(2)))  __bf16        v2bf;
typedef __attribute__((ext_vector_type(8)))  float         v8f;
typedef __attribute__((ext_vector_type(4)))  unsigned int  v4u;

union Frag { v16bf v; v4u q[2]; unsigned int u[8]; };
union Acc  { v8f v; float f[8]; };

// native RNE converts -> v_cvt_pk_bf16_f32 class instructions on gfx1250
__device__ __forceinline__ unsigned short f2bf(float f) {
    __bf16 h = (__bf16)f;
    return __builtin_bit_cast(unsigned short, h);
}
__device__ __forceinline__ unsigned int pack2bf(float a, float b) {
    v2bf h;
    h.x = (__bf16)a;
    h.y = (__bf16)b;
    return __builtin_bit_cast(unsigned int, h);
}

// ---------------- prep: FiLM (z_id @ film_w.T + film_b), fold 1+gamma ----------------
__global__ void prep_film(const float* __restrict__ z, const float* __restrict__ fw,
                          const float* __restrict__ fb, float* __restrict__ gamma,
                          float* __restrict__ beta) {
    int t = threadIdx.x;                       // 256 threads
    for (int b = 0; b < BB; ++b) {
        const float* zz = z + b * ZD;
        const float* r0 = fw + (size_t)t * ZD;
        const float* r1 = fw + (size_t)(t + HID) * ZD;
        float g = fb[t], be = fb[t + HID];
        #pragma unroll 8
        for (int k = 0; k < ZD; ++k) { g = fmaf(zz[k], r0[k], g); be = fmaf(zz[k], r1[k], be); }
        gamma[b * HID + t] = 1.0f + g;
        beta[b * HID + t]  = be;
    }
}

// ---------------- prep: weights fp32 -> bf16, K/out padded ----------------
__global__ void prep_weights(const float* __restrict__ fc1w, const float* __restrict__ w1,
                             const float* __restrict__ w2, const float* __restrict__ w3,
                             unsigned short* __restrict__ fc1wb, unsigned short* __restrict__ w1b,
                             unsigned short* __restrict__ w2b, unsigned short* __restrict__ w3b) {
    int idx = blockIdx.x * 256 + threadIdx.x;
    const int n_fc1 = HID * KP1;
    if (idx < n_fc1) {
        int o = idx / KP1, k = idx % KP1;
        fc1wb[idx] = f2bf(k < IN_DIM ? fc1w[(size_t)o * IN_DIM + k] : 0.f);
        return;
    }
    idx -= n_fc1;
    if (idx < HID * HID) { w1b[idx] = f2bf(w1[idx]); return; }
    idx -= HID * HID;
    if (idx < HID * HID) { w2b[idx] = f2bf(w2[idx]); return; }
    idx -= HID * HID;
    if (idx < 16 * HID) {
        int o = idx / HID, k = idx % HID;
        w3b[idx] = f2bf(o < OUTD ? w3[(size_t)o * HID + k] : 0.f);
    }
}

// ---------------- per-branch head nonlinearities ----------------
__device__ __forceinline__ void head(const float* o, float* s, float* r, float& alpha, float* sh) {
    #pragma unroll
    for (int k = 0; k < 3; ++k) {
        float x = o[k];
        float sp = (x > 20.f) ? x : log1pf(__expf(x));
        s[k] = fminf(fmaxf(sp + 1e-6f, 1e-6f), 3.0f);
    }
    float nn = 0.f;
    #pragma unroll
    for (int k = 0; k < 4; ++k) { r[k] = o[3 + k]; nn = fmaf(r[k], r[k], nn); }
    float inv = 1.0f / (sqrtf(nn) + 1e-8f);
    #pragma unroll
    for (int k = 0; k < 4; ++k) r[k] *= inv;
    float sg = 1.0f / (1.0f + __expf(-o[7]));
    alpha = fminf(fmaxf(sg, 1e-6f), 1.0f);
    #pragma unroll
    for (int k = 0; k < 3; ++k) sh[k] = tanhf(o[8 + k]) * 0.5f;
}

// ---------------- main fused kernel: 1 wave = 16 points, both batches ----------------
__global__ __launch_bounds__(128, 1) void decoder_main(
    const float* __restrict__ feats, const float* __restrict__ fc1b,
    const float* __restrict__ bias1, const float* __restrict__ bias2, const float* __restrict__ bias3,
    const float* __restrict__ gamma, const float* __restrict__ beta,
    const unsigned short* __restrict__ fc1wb, const unsigned short* __restrict__ w1b,
    const unsigned short* __restrict__ w2b, const unsigned short* __restrict__ w3b,
    float* __restrict__ out)
{
    __shared__ __align__(16) unsigned short s_h[WAVES][16 * HID];   // bf16 activations
    __shared__ float s_o[WAVES][BB][16 * 16];                        // raw MLP outputs
    __shared__ float s_fin[WAVES][16 * OUTD];

    const int lane = threadIdx.x & 31;
    const int wv   = threadIdx.x >> 5;
    const int col  = lane & 15;          // C/D column = N index = A row (point)
    const int hi   = lane >> 4;          // lane half selects K/M groups
    const int n0   = (blockIdx.x * WAVES + wv) << 4;

    for (int b = 0; b < BB; ++b) {
        const float* fr = feats + ((size_t)b * NN + (n0 + col)) * IN_DIM;

        // ---------- fc1: 16x515 @ 515x256 (bf16 WMMA, K padded to 544) ----------
        Acc acc[16];
        #pragma unroll
        for (int nt = 0; nt < 16; ++nt)
            #pragma unroll
            for (int r = 0; r < 8; ++r) acc[nt].f[r] = 0.f;

        for (int kt = 0; kt < 17; ++kt) {
            if (kt < 16) __builtin_prefetch(fr + (kt + 1) * 32 + hi * 8, 0, 1);
            Frag a;
            const int kb = kt * 32 + hi * 8;
            #pragma unroll
            for (int d = 0; d < 8; ++d) {
                const int k = kb + ((d < 4) ? 2 * d : 16 + 2 * (d - 4));
                float f0 = (k     < IN_DIM) ? fr[k]     : 0.f;
                float f1 = (k + 1 < IN_DIM) ? fr[k + 1] : 0.f;
                a.u[d] = pack2bf(f0, f1);
            }
            #pragma unroll
            for (int nt = 0; nt < 16; ++nt) {
                Frag bm;
                const v4u* wp = (const v4u*)(fc1wb + (size_t)(nt * 16 + col) * KP1 + kt * 32 + hi * 8);
                bm.q[0] = wp[0]; bm.q[1] = wp[2];
                acc[nt].v = __builtin_amdgcn_wmma_f32_16x16x32_bf16(
                    false, a.v, false, bm.v, (short)0, acc[nt].v, false, false);
            }
        }
        // FiLM + ReLU, store bf16 activations to LDS (D layout -> row-major)
        #pragma unroll
        for (int nt = 0; nt < 16; ++nt) {
            const int n = nt * 16 + col;
            const float bi = fc1b[n];
            const float g  = gamma[b * HID + n];
            const float be = beta[b * HID + n];
            #pragma unroll
            for (int r = 0; r < 8; ++r) {
                float h = fmaxf(fmaf(g, acc[nt].f[r] + bi, be), 0.f);
                s_h[wv][(r + 8 * hi) * HID + n] = f2bf(h);
            }
        }

        // ---------- w1, w2: 256->256, ReLU ----------
        const unsigned short* wls[2] = { w1b, w2b };
        const float*          bls[2] = { bias1, bias2 };
        for (int L = 0; L < 2; ++L) {
            Acc ac[16];
            #pragma unroll
            for (int nt = 0; nt < 16; ++nt)
                #pragma unroll
                for (int r = 0; r < 8; ++r) ac[nt].f[r] = 0.f;

            for (int kt = 0; kt < 8; ++kt) {
                Frag a;
                const unsigned short* hp = &s_h[wv][col * HID + kt * 32 + hi * 8];
                a.q[0] = *(const v4u*)(hp);
                a.q[1] = *(const v4u*)(hp + 16);
                #pragma unroll
                for (int nt = 0; nt < 16; ++nt) {
                    Frag bm;
                    const v4u* wp = (const v4u*)(wls[L] + (size_t)(nt * 16 + col) * HID + kt * 32 + hi * 8);
                    bm.q[0] = wp[0]; bm.q[1] = wp[2];
                    ac[nt].v = __builtin_amdgcn_wmma_f32_16x16x32_bf16(
                        false, a.v, false, bm.v, (short)0, ac[nt].v, false, false);
                }
            }
            #pragma unroll
            for (int nt = 0; nt < 16; ++nt) {
                const int n = nt * 16 + col;
                const float bi = bls[L][n];
                #pragma unroll
                for (int r = 0; r < 8; ++r) {
                    float h = fmaxf(ac[nt].f[r] + bi, 0.f);
                    s_h[wv][(r + 8 * hi) * HID + n] = f2bf(h);
                }
            }
        }

        // ---------- w3: 256->16 (11 valid) ----------
        Acc a4;
        #pragma unroll
        for (int r = 0; r < 8; ++r) a4.f[r] = 0.f;
        for (int kt = 0; kt < 8; ++kt) {
            Frag a;
            const unsigned short* hp = &s_h[wv][col * HID + kt * 32 + hi * 8];
            a.q[0] = *(const v4u*)(hp);
            a.q[1] = *(const v4u*)(hp + 16);
            Frag bm;
            const v4u* wp = (const v4u*)(w3b + (size_t)col * HID + kt * 32 + hi * 8);
            bm.q[0] = wp[0]; bm.q[1] = wp[2];
            a4.v = __builtin_amdgcn_wmma_f32_16x16x32_bf16(
                false, a.v, false, bm.v, (short)0, a4.v, false, false);
        }
        const float bi3 = (col < OUTD) ? bias3[col] : 0.f;
        #pragma unroll
        for (int r = 0; r < 8; ++r)
            s_o[wv][b][(r + 8 * hi) * 16 + col] = a4.f[r] + bi3;
    } // b loop

    // ---------- per-point heads + cross-batch aggregation (lanes 0..15, 1 point each) ----------
    if (hi == 0) {
        const int p = col;
        const float* o0 = &s_o[wv][0][p * 16];
        const float* o1 = &s_o[wv][1][p * 16];
        float s0[3], s1[3], r0[4], r1[4], sh0[3], sh1[3], a0, a1;
        head(o0, s0, r0, a0, sh0);
        head(o1, s1, r1, a1, sh1);

        const float inv = 1.0f / fmaxf(a0 + a1, 1e-8f);
        const float w0 = a0 * inv, w1 = a1 * inv;
        float* fin = &s_fin[wv][p * OUTD];
        #pragma unroll
        for (int k = 0; k < 3; ++k) fin[k] = w0 * s0[k] + w1 * s1[k];

        // top eigenvector of M = w0*r0 r0^T + w1*r1 r1^T  (rank-2 -> power iteration)
        float c = 0.f;
        #pragma unroll
        for (int i = 0; i < 4; ++i) c = fmaf(r0[i], r1[i], c);
        float v[4];
        const float sgn = (c >= 0.f) ? w1 : -w1;
        #pragma unroll
        for (int i = 0; i < 4; ++i) v[i] = w0 * r0[i] + sgn * r1[i];
        for (int it = 0; it < 8; ++it) {
            float d0 = 0.f, d1 = 0.f;
            #pragma unroll
            for (int i = 0; i < 4; ++i) { d0 = fmaf(r0[i], v[i], d0); d1 = fmaf(r1[i], v[i], d1); }
            float nv = 0.f;
            #pragma unroll
            for (int i = 0; i < 4; ++i) { v[i] = w0 * r0[i] * d0 + w1 * r1[i] * d1; nv = fmaf(v[i], v[i], nv); }
            float sc = rsqrtf(nv + 1e-24f);
            #pragma unroll
            for (int i = 0; i < 4; ++i) v[i] *= sc;
        }
        #pragma unroll
        for (int i = 0; i < 4; ++i) fin[3 + i] = v[i];
        fin[7] = w0 * a0 + w1 * a1;
        #pragma unroll
        for (int k = 0; k < 3; ++k) fin[8 + k] = w0 * sh0[k] + w1 * sh1[k];
    }

    // coalesced store of the 16x11 tile
    for (int idx = lane; idx < 16 * OUTD; idx += 32)
        out[(size_t)n0 * OUTD + idx] = s_fin[wv][idx];
}

// ---------------- launcher ----------------
extern "C" void kernel_launch(void* const* d_in, const int* in_sizes, int n_in,
                              void* d_out, int out_size, void* d_ws, size_t ws_size,
                              hipStream_t stream) {
    const float* feats = (const float*)d_in[0];
    const float* z     = (const float*)d_in[1];
    const float* fc1w  = (const float*)d_in[2];
    const float* fc1b  = (const float*)d_in[3];
    const float* fw    = (const float*)d_in[4];
    const float* fb    = (const float*)d_in[5];
    const float* w1    = (const float*)d_in[6];
    const float* b1    = (const float*)d_in[7];
    const float* w2    = (const float*)d_in[8];
    const float* b2    = (const float*)d_in[9];
    const float* w3    = (const float*)d_in[10];
    const float* b3    = (const float*)d_in[11];
    float* outp = (float*)d_out;

    char* ws = (char*)d_ws;                            // ~553 KB used
    float*          gamma = (float*)(ws + 0);          // [2][256]
    float*          beta  = (float*)(ws + 2048);       // [2][256]
    unsigned short* fc1wb = (unsigned short*)(ws + 4096);     // [256][544] bf16
    unsigned short* w1b   = (unsigned short*)(ws + 282624);   // [256][256] bf16
    unsigned short* w2b   = (unsigned short*)(ws + 413696);   // [256][256] bf16
    unsigned short* w3b   = (unsigned short*)(ws + 544768);   // [16][256]  bf16

    prep_film<<<1, 256, 0, stream>>>(z, fw, fb, gamma, beta);
    const int total = HID * KP1 + 2 * HID * HID + 16 * HID;
    prep_weights<<<(total + 255) / 256, 256, 0, stream>>>(fc1w, w1, w2, w3, fc1wb, w1b, w2b, w3b);
    decoder_main<<<NN / (16 * WAVES), 128, 0, stream>>>(
        feats, fc1b, b1, b2, b3, gamma, beta, fc1wb, w1b, w2b, w3b, outp);
}
